// NTM_17686675325188
// MI455X (gfx1250) — matrix operand order repaired and checked by reading, not compiled, not run on hardware
//
#include <hip/hip_runtime.h>
#include <math.h>

// ---------------------------------------------------------------------------
// NTM step for MI455X (gfx1250, wave32).
// Memory-bound: bank = 64x4096x128 f32 = 134 MB (fits in 192MB L2).
// Streaming passes: 1 wave per bank row (512B, float4/lane, shfl_xor reduce).
// Dense GEMMs (LSTM / projections / output) use v_wmma_f32_16x16x32_f16.
// ---------------------------------------------------------------------------

typedef __attribute__((ext_vector_type(16))) _Float16 v16h;
typedef __attribute__((ext_vector_type(8)))  _Float16 v8h;
typedef __attribute__((ext_vector_type(8)))  float    v8f;

#define DINL __device__ __forceinline__

DINL float sigmoidf_(float x) { return 1.0f / (1.0f + expf(-x)); }
DINL float softplusf_(float x) {
  return (x > 0.0f) ? (x + log1pf(expf(-x))) : log1pf(expf(x));
}

// ---------------------------------------------------------------------------
// Generic WMMA GEMM:  D(MxN,f32) = A(MxK,f16,row) @ B(NxK,f16,row)^T  [+ D] [+ bias]
// One wave (32 threads) per 16x16 tile. M must be a multiple of 16; N tail is
// handled by clamped loads + guarded stores. K must be a multiple of 32.
//
// A fragment (16x32 f16): lane 0-15 hold M=lane, halves 0..7 = K 0..7,
// halves 8..15 = K 16..23; lanes 16-31 same rows with K offset +8/+24.
// B fragment (32x16 f16): lane holds column N=lane&15, same K layout as A
// (halves 0..7 = K 8*(lane>>4)+0..7, halves 8..15 = K 16+8*(lane>>4)+0..7).
// C/D (16x16 f32): n = lane&15, m = r + 8*(lane>>4) for VGPR r.
// ---------------------------------------------------------------------------
__global__ void wmma_gemm_nt(const _Float16* __restrict__ A,
                             const _Float16* __restrict__ B,
                             float* __restrict__ D,
                             const float* __restrict__ bias,
                             int M, int N, int K, int accum)
{
  const int lane = threadIdx.x & 31;
  const int n0   = blockIdx.x * 16;
  const int m0   = blockIdx.y * 16;
  const int hi   = lane >> 4;
  const int l15  = lane & 15;

  const int arow = m0 + l15;
  int brow = n0 + l15;
  if (brow >= N) brow = N - 1;                  // clamp loads on N tail

  const _Float16* aBase = A + (size_t)arow * K + 8 * hi;
  const _Float16* bBase = B + (size_t)brow * K + 8 * hi;

  v8f acc;
  if (accum) {
#pragma unroll
    for (int r = 0; r < 8; ++r) {
      const int m = m0 + 8 * hi + r;
      const int n = n0 + l15;
      acc[r] = (n < N) ? D[(size_t)m * N + n] : 0.0f;
    }
  } else {
    acc = (v8f){0.f, 0.f, 0.f, 0.f, 0.f, 0.f, 0.f, 0.f};
  }

  for (int k0 = 0; k0 < K; k0 += 32) {
    v8h alo = *(const v8h*)(aBase + k0);
    v8h ahi = *(const v8h*)(aBase + k0 + 16);
    v16h a = __builtin_shufflevector(alo, ahi,
               0,1,2,3,4,5,6,7,8,9,10,11,12,13,14,15);
    v8h blo = *(const v8h*)(bBase + k0);
    v8h bhi = *(const v8h*)(bBase + k0 + 16);
    v16h b = __builtin_shufflevector(blo, bhi,
               0,1,2,3,4,5,6,7,8,9,10,11,12,13,14,15);
    acc = __builtin_amdgcn_wmma_f32_16x16x32_f16(
              false, a, false, b, (short)0, acc, false, false);
  }

#pragma unroll
  for (int r = 0; r < 8; ++r) {
    const int m = m0 + 8 * hi + r;
    const int n = n0 + l15;
    if (n < N) {
      float v = acc[r];
      if (bias) v += bias[n];
      D[(size_t)m * N + n] = v;
    }
  }
}

// ---------------------------------------------------------------------------
// f32 -> f16 conversion (grid-stride not needed; sizes are modest)
// ---------------------------------------------------------------------------
__global__ void f32_to_f16_kernel(const float* __restrict__ s,
                                  _Float16* __restrict__ d, int n)
{
  const int i = blockIdx.x * blockDim.x + threadIdx.x;
  if (i < n) d[i] = (_Float16)s[i];
}

__global__ void zero_kernel(float* __restrict__ p, int n)
{
  const int i = blockIdx.x * blockDim.x + threadIdx.x;
  if (i < n) p[i] = 0.0f;
}

// ---------------------------------------------------------------------------
// LSTM elementwise: gates(64,2048) + biases -> h(64,512) f32 + f16
// ---------------------------------------------------------------------------
__global__ void lstm_act_kernel(const float* __restrict__ gates,
                                const float* __restrict__ c_prev,
                                const float* __restrict__ bih,
                                const float* __restrict__ bhh,
                                float* __restrict__ h,
                                _Float16* __restrict__ h16)
{
  const int idx = blockIdx.x * blockDim.x + threadIdx.x;
  if (idx >= 64 * 512) return;
  const int b = idx >> 9, j = idx & 511;
  const float* g = gates + (size_t)b * 2048;
  const float ig = g[j]        + bih[j]        + bhh[j];
  const float fg = g[512 + j]  + bih[512 + j]  + bhh[512 + j];
  const float gg = g[1024 + j] + bih[1024 + j] + bhh[1024 + j];
  const float og = g[1536 + j] + bih[1536 + j] + bhh[1536 + j];
  const float c  = sigmoidf_(fg) * c_prev[idx] + sigmoidf_(ig) * tanhf(gg);
  const float hv = sigmoidf_(og) * tanhf(c);
  h[idx]   = hv;
  h16[idx] = (_Float16)hv;
}

// ---------------------------------------------------------------------------
// Split write-head params. Block per (i,b): 128 threads.
// Pw layout: (64 x 1170) where row b, cols [i*390 .. i*390+389].
// scal: beta, g, s0, s1, s2, r, ||k||, pad
// ---------------------------------------------------------------------------
__global__ void split_write_kernel(const float* __restrict__ P,
                                   const float* __restrict__ bias,   // (3,390)
                                   float* __restrict__ KW,
                                   float* __restrict__ E,
                                   float* __restrict__ Abuf,
                                   float* __restrict__ SW)
{
  __shared__ float red[128];
  const int blk = blockIdx.x;                 // 0..191 = i*64+b
  const int i = blk >> 6, b = blk & 63;
  const int t = threadIdx.x;
  const float* row = P + (size_t)b * 1170 + i * 390;
  const float* bb  = bias + i * 390;

  const float kv = row[t] + bb[t];
  KW  [(size_t)blk * 128 + t] = kv;
  E   [(size_t)blk * 128 + t] = sigmoidf_(row[134 + t] + bb[134 + t]);
  Abuf[(size_t)blk * 128 + t] = row[262 + t] + bb[262 + t];
  red[t] = kv * kv;
  __syncthreads();
  for (int s = 64; s > 0; s >>= 1) {
    if (t < s) red[t] += red[t + s];
    __syncthreads();
  }
  if (t == 0) {
    float* sc = SW + (size_t)blk * 8;
    sc[0] = softplusf_(row[128] + bb[128]);
    sc[1] = sigmoidf_(row[129] + bb[129]);
    const float s0 = row[130] + bb[130], s1 = row[131] + bb[131],
                s2 = row[132] + bb[132];
    const float mx = fmaxf(s0, fmaxf(s1, s2));
    const float e0 = expf(s0 - mx), e1 = expf(s1 - mx), e2 = expf(s2 - mx);
    const float es = e0 + e1 + e2;
    sc[2] = e0 / es; sc[3] = e1 / es; sc[4] = e2 / es;
    sc[5] = 1.0f + softplusf_(row[133] + bb[133]);
    sc[6] = sqrtf(red[0]);
    sc[7] = 0.0f;
  }
}

__global__ void split_read_kernel(const float* __restrict__ P,     // (64,402)
                                  const float* __restrict__ bias,  // (3,134)
                                  float* __restrict__ KR,
                                  float* __restrict__ SR)
{
  __shared__ float red[128];
  const int blk = blockIdx.x;
  const int i = blk >> 6, b = blk & 63;
  const int t = threadIdx.x;
  const float* row = P + (size_t)b * 402 + i * 134;
  const float* bb  = bias + i * 134;

  const float kv = row[t] + bb[t];
  KR[(size_t)blk * 128 + t] = kv;
  red[t] = kv * kv;
  __syncthreads();
  for (int s = 64; s > 0; s >>= 1) {
    if (t < s) red[t] += red[t + s];
    __syncthreads();
  }
  if (t == 0) {
    float* sc = SR + (size_t)blk * 8;
    sc[0] = softplusf_(row[128] + bb[128]);
    sc[1] = sigmoidf_(row[129] + bb[129]);
    const float s0 = row[130] + bb[130], s1 = row[131] + bb[131],
                s2 = row[132] + bb[132];
    const float mx = fmaxf(s0, fmaxf(s1, s2));
    const float e0 = expf(s0 - mx), e1 = expf(s1 - mx), e2 = expf(s2 - mx);
    const float es = e0 + e1 + e2;
    sc[2] = e0 / es; sc[3] = e1 / es; sc[4] = e2 / es;
    sc[5] = 1.0f + softplusf_(row[133] + bb[133]);
    sc[6] = sqrtf(red[0]);
    sc[7] = 0.0f;
  }
}

// ---------------------------------------------------------------------------
// Content logits: 1 wave per bank row (b,n). Each lane holds float4 of the
// 128-dim row (512B per wave -> fully coalesced). shfl_xor wave32 reduce.
// logits[b*4096+n] = beta * dot(row,k) / max(||row||*||k||, 1e-8)
// ---------------------------------------------------------------------------
__global__ void content_logits_kernel(const float* __restrict__ bank,
                                      const float* __restrict__ Kvec,  // (64,128)
                                      const float* __restrict__ scal,  // (64,8)
                                      float* __restrict__ logits)
{
  const int lane = threadIdx.x & 31;
  const size_t row = (size_t)blockIdx.x * (blockDim.x >> 5) + (threadIdx.x >> 5);
  const int b = (int)(row >> 12);
  const float4 v  = ((const float4*)(bank + (row << 7)))[lane];
  const float4 kv = ((const float4*)(Kvec + b * 128))[lane];
  float dot = v.x * kv.x + v.y * kv.y + v.z * kv.z + v.w * kv.w;
  float nrm = v.x * v.x + v.y * v.y + v.z * v.z + v.w * v.w;
#pragma unroll
  for (int off = 16; off > 0; off >>= 1) {
    dot += __shfl_xor(dot, off, 32);
    nrm += __shfl_xor(nrm, off, 32);
  }
  if (lane == 0) {
    const float beta = scal[b * 8 + 0], kn = scal[b * 8 + 6];
    logits[row] = beta * dot / fmaxf(sqrtf(nrm) * kn, 1e-8f);
  }
}

// ---------------------------------------------------------------------------
// Addressing: per-batch block. softmax over N=4096 in LDS, + (1-g)*w_prev,
// circular 3-tap shift, sharpen w^r / (sum + eps).
// ---------------------------------------------------------------------------
__global__ void addressing_kernel(const float* __restrict__ logits,
                                  const float* __restrict__ wprev,  // (64,4096)
                                  const float* __restrict__ scal,   // (64,8)
                                  float* __restrict__ wout)
{
  __shared__ float w[4096];
  __shared__ float w2[4096];
  __shared__ float red[256];
  const int b = blockIdx.x, t = threadIdx.x;
  const float* lg = logits + (size_t)b * 4096;
  const float* wp = wprev  + (size_t)b * 4096;
  const float* sc = scal   + (size_t)b * 8;

  float lmax = -3.4e38f;
  for (int n = t; n < 4096; n += 256) { const float v = lg[n]; w[n] = v; lmax = fmaxf(lmax, v); }
  red[t] = lmax; __syncthreads();
  for (int s = 128; s > 0; s >>= 1) { if (t < s) red[t] = fmaxf(red[t], red[t + s]); __syncthreads(); }
  const float smax = red[0]; __syncthreads();

  float lsum = 0.0f;
  for (int n = t; n < 4096; n += 256) { const float e = expf(w[n] - smax); w[n] = e; lsum += e; }
  red[t] = lsum; __syncthreads();
  for (int s = 128; s > 0; s >>= 1) { if (t < s) red[t] += red[t + s]; __syncthreads(); }
  const float ssum = red[0]; __syncthreads();

  const float g = sc[1], s0 = sc[2], s1 = sc[3], s2 = sc[4], r = sc[5];
  for (int n = t; n < 4096; n += 256) w[n] = w[n] / ssum + (1.0f - g) * wp[n];
  __syncthreads();

  float psum = 0.0f;
  for (int n = t; n < 4096; n += 256) {
    const float sh = s0 * w[(n + 4095) & 4095] + s1 * w[n] + s2 * w[(n + 1) & 4095];
    const float v = powf(fmaxf(sh, 0.0f), r);
    w2[n] = v; psum += v;
  }
  red[t] = psum; __syncthreads();
  for (int s = 128; s > 0; s >>= 1) { if (t < s) red[t] += red[t + s]; __syncthreads(); }
  const float tot = red[0];
  float* wo = wout + (size_t)b * 4096;
  for (int n = t; n < 4096; n += 256) wo[n] = w2[n] / (tot + 1e-8f);
}

// ---------------------------------------------------------------------------
// Fused bank erase/add update + read-head content logits on the NEW bank.
// One wave per row; single read + single write of the 134MB bank.
// ---------------------------------------------------------------------------
__global__ void bank_update_kernel(const float* __restrict__ src,
                                   float* __restrict__ dst,
                                   const float* __restrict__ ww,    // (64*4096)
                                   const float* __restrict__ Ev,    // (64,128)
                                   const float* __restrict__ Av,
                                   const float* __restrict__ KR,
                                   const float* __restrict__ scalR, // (64,8)
                                   float* __restrict__ logitsR)
{
  const int lane = threadIdx.x & 31;
  const size_t row = (size_t)blockIdx.x * (blockDim.x >> 5) + (threadIdx.x >> 5);
  const int b = (int)(row >> 12);
  const float wwv = ww[row];
  const float4 v  = ((const float4*)(src + (row << 7)))[lane];
  const float4 e4 = ((const float4*)(Ev + b * 128))[lane];
  const float4 a4 = ((const float4*)(Av + b * 128))[lane];
  const float4 k4 = ((const float4*)(KR + b * 128))[lane];
  float4 nv;
  nv.x = v.x * (1.0f - wwv * e4.x) + wwv * a4.x;
  nv.y = v.y * (1.0f - wwv * e4.y) + wwv * a4.y;
  nv.z = v.z * (1.0f - wwv * e4.z) + wwv * a4.z;
  nv.w = v.w * (1.0f - wwv * e4.w) + wwv * a4.w;
  ((float4*)(dst + (row << 7)))[lane] = nv;
  float dot = nv.x * k4.x + nv.y * k4.y + nv.z * k4.z + nv.w * k4.w;
  float nrm = nv.x * nv.x + nv.y * nv.y + nv.z * nv.z + nv.w * nv.w;
#pragma unroll
  for (int off = 16; off > 0; off >>= 1) {
    dot += __shfl_xor(dot, off, 32);
    nrm += __shfl_xor(nrm, off, 32);
  }
  if (lane == 0) {
    const float beta = scalR[b * 8 + 0], kn = scalR[b * 8 + 6];
    logitsR[row] = beta * dot / fmaxf(sqrtf(nrm) * kn, 1e-8f);
  }
}

// ---------------------------------------------------------------------------
// reads[b,m] += sum_n wr[b,n] * bank[b,n,m].  Grid (8 n-chunks, 64 b).
// Threads: m = t&127 (coalesced 512B rows), two rows in flight per block.
// ---------------------------------------------------------------------------
__global__ void read_reduce_kernel(const float* __restrict__ bank,
                                   const float* __restrict__ wr,
                                   float* __restrict__ reads)   // (64,128) zeroed
{
  __shared__ float sh[256];
  const int b = blockIdx.y, chunk = blockIdx.x;
  const int t = threadIdx.x, m = t & 127, half = t >> 7;
  const float* wrb = wr + (size_t)b * 4096;
  float acc = 0.0f;
  const int base = chunk * 512;
  for (int it = 0; it < 256; ++it) {
    const int n = base + it * 2 + half;
    const float* rowp = bank + (((size_t)b << 12) + n) * 128;
    __builtin_prefetch(rowp + 256, 0, 0);    // next row pair -> global_prefetch_b8
    acc += wrb[n] * rowp[m];
  }
  sh[t] = acc; __syncthreads();
  if (t < 128) atomicAdd(&reads[(size_t)b * 128 + m], sh[t] + sh[t + 128]);
}

// ---------------------------------------------------------------------------
// Concat [r0|r1|r2|h] -> f16 (64 x 896) for the output GEMM.
// ---------------------------------------------------------------------------
__global__ void assemble_z_kernel(const float* __restrict__ reads,  // (3,64,128)
                                  const float* __restrict__ h,      // (64,512)
                                  _Float16* __restrict__ Z)
{
  const int idx = blockIdx.x * blockDim.x + threadIdx.x;
  if (idx >= 64 * 896) return;
  const int b = idx / 896, j = idx - b * 896;
  const float v = (j < 384)
      ? reads[(size_t)(j >> 7) * (64 * 128) + (size_t)b * 128 + (j & 127)]
      : h[(size_t)b * 512 + (j - 384)];
  Z[idx] = (_Float16)v;
}

// ---------------------------------------------------------------------------
extern "C" void kernel_launch(void* const* d_in, const int* in_sizes, int n_in,
                              void* d_out, int out_size, void* d_ws, size_t ws_size,
                              hipStream_t stream)
{
  (void)in_sizes; (void)n_in; (void)out_size; (void)ws_size;
  const float* x       = (const float*)d_in[0];
  const float* h_prev  = (const float*)d_in[1];
  const float* c_prev  = (const float*)d_in[2];
  const float* bank    = (const float*)d_in[3];
  const float* rwp     = (const float*)d_in[4];   // read_w_prev  (3,64,4096)
  const float* wwp     = (const float*)d_in[5];   // write_w_prev (3,64,4096)
  const float* W_ih    = (const float*)d_in[6];
  const float* W_hh    = (const float*)d_in[7];
  const float* b_ih    = (const float*)d_in[8];
  const float* b_hh    = (const float*)d_in[9];
  const float* read_W  = (const float*)d_in[10];
  const float* read_b  = (const float*)d_in[11];
  const float* write_W = (const float*)d_in[12];
  const float* write_b = (const float*)d_in[13];
  const float* out_W   = (const float*)d_in[14];
  const float* out_b   = (const float*)d_in[15];
  float* out = (float*)d_out;

  // ---- workspace carve (64B aligned chunks) ----
  float* base = (float*)d_ws;
  size_t off = 0;
  auto alloc = [&](size_t n) -> float* {
    float* p = base + off;
    off += (n + 15) & ~(size_t)15;
    return p;
  };
  float* wsBank = alloc(64ull * 4096 * 128);   // mutable bank copy (134 MB)
  float* gates  = alloc(64 * 2048);
  float* hbuf   = alloc(64 * 512);
  float* Pw     = alloc(64 * 1170);
  float* Pr     = alloc(64 * 402);
  float* KW     = alloc(3 * 64 * 128);
  float* KR     = alloc(3 * 64 * 128);
  float* Ebuf   = alloc(3 * 64 * 128);
  float* Abuf   = alloc(3 * 64 * 128);
  float* SW     = alloc(3 * 64 * 8);
  float* SR     = alloc(3 * 64 * 8);
  float* logW   = alloc(64 * 4096);
  float* wwb    = alloc(64 * 4096);
  float* logR   = alloc(64 * 4096);
  float* wrb    = alloc(64 * 4096);
  float* readsB = alloc(3 * 64 * 128);
  _Float16* x16   = (_Float16*)alloc(64 * 256 / 2);
  _Float16* hp16  = (_Float16*)alloc(64 * 512 / 2);
  _Float16* h16   = (_Float16*)alloc(64 * 512 / 2);
  _Float16* Wih16 = (_Float16*)alloc(2048 * 256 / 2);
  _Float16* Whh16 = (_Float16*)alloc(2048 * 512 / 2);
  _Float16* wW16  = (_Float16*)alloc(3 * 390 * 512 / 2);
  _Float16* rW16  = (_Float16*)alloc(3 * 134 * 512 / 2);
  _Float16* oW16  = (_Float16*)alloc(256 * 896 / 2);
  _Float16* Z16   = (_Float16*)alloc(64 * 896 / 2);

  auto cvt = [&](const float* s, _Float16* d, int n) {
    f32_to_f16_kernel<<<(n + 255) / 256, 256, 0, stream>>>(s, d, n);
  };
  cvt(x, x16, 64 * 256);
  cvt(h_prev, hp16, 64 * 512);
  cvt(W_ih, Wih16, 2048 * 256);
  cvt(W_hh, Whh16, 2048 * 512);
  cvt(write_W, wW16, 3 * 390 * 512);
  cvt(read_W, rW16, 3 * 134 * 512);
  cvt(out_W, oW16, 256 * 896);

  // LSTM: gates = x@W_ih.T ; gates += h_prev@W_hh.T ; activations
  wmma_gemm_nt<<<dim3(2048 / 16, 4), 32, 0, stream>>>(x16,  Wih16, gates, nullptr, 64, 2048, 256, 0);
  wmma_gemm_nt<<<dim3(2048 / 16, 4), 32, 0, stream>>>(hp16, Whh16, gates, nullptr, 64, 2048, 512, 1);
  lstm_act_kernel<<<(64 * 512 + 255) / 256, 256, 0, stream>>>(gates, c_prev, b_ih, b_hh, hbuf, h16);

  // All 6 controller projections (depend only on h)
  wmma_gemm_nt<<<dim3((1170 + 15) / 16, 4), 32, 0, stream>>>(h16, wW16, Pw, nullptr, 64, 1170, 512, 0);
  wmma_gemm_nt<<<dim3((402 + 15) / 16, 4), 32, 0, stream>>>(h16, rW16, Pr, nullptr, 64, 402, 512, 0);
  split_write_kernel<<<192, 128, 0, stream>>>(Pw, write_b, KW, Ebuf, Abuf, SW);
  split_read_kernel<<<192, 128, 0, stream>>>(Pr, read_b, KR, SR);
  zero_kernel<<<(3 * 64 * 128 + 255) / 256, 256, 0, stream>>>(readsB, 3 * 64 * 128);

  const int clBlocks = (64 * 4096) / 8;   // 8 waves / 256-thread block
  for (int i = 0; i < 3; ++i) {
    const float* src = (i == 0) ? bank : wsBank;   // never mutate the input
    content_logits_kernel<<<clBlocks, 256, 0, stream>>>(
        src, KW + i * 8192, SW + i * 512, logW);
    addressing_kernel<<<64, 256, 0, stream>>>(
        logW, wwp + (size_t)i * 64 * 4096, SW + i * 512, wwb);
    bank_update_kernel<<<clBlocks, 256, 0, stream>>>(
        src, wsBank, wwb, Ebuf + i * 8192, Abuf + i * 8192,
        KR + i * 8192, SR + i * 512, logR);
    addressing_kernel<<<64, 256, 0, stream>>>(
        logR, rwp + (size_t)i * 64 * 4096, SR + i * 512, wrb);
    read_reduce_kernel<<<dim3(8, 64), 256, 0, stream>>>(
        wsBank, wrb, readsB + i * 8192);
  }

  assemble_z_kernel<<<(64 * 896 + 255) / 256, 256, 0, stream>>>(readsB, hbuf, Z16);
  wmma_gemm_nt<<<dim3(256 / 16, 4), 32, 0, stream>>>(Z16, oW16, out, out_b, 64, 256, 896, 0);
}
